// LlamaAttention_7670811591232
// MI455X (gfx1250) — matrix operand order repaired and checked by reading, not compiled, not run on hardware
//
#include <hip/hip_runtime.h>
#include <stdint.h>

#define S_LEN    2048
#define HIDDEN   4096
#define N_HEADS  32
#define N_KV     8
#define HEAD_DIM 128

typedef __attribute__((ext_vector_type(16))) __bf16 v16bf;
typedef __attribute__((ext_vector_type(8)))  float  v8f;
typedef __attribute__((ext_vector_type(4)))  unsigned int v4u;
typedef __attribute__((ext_vector_type(8)))  int    v8i;
typedef __attribute__((ext_vector_type(4)))  int    v4i;

union Frag { unsigned int u[8]; v16bf v; };
union CAcc { v8f v; float f[8]; };

__device__ __forceinline__ unsigned short f2bf(float f) {
  unsigned int x = __float_as_uint(f);
  x += 0x7FFFu + ((x >> 16) & 1u);       // round-to-nearest-even
  return (unsigned short)(x >> 16);
}

// ---------------------------------------------------------------------------
// GEMM: C[M,N] = A[M,K] (fp32) x W[K,N] (fp32) + bias, bf16 WMMA, fp32 accum.
// Block tile 128x128, K-step 32. 8 waves: 2(m) x 4(n); wave tile 64x32.
// ---------------------------------------------------------------------------
#define BM 128
#define BN 128
#define BK 32
#define LDT 34   // padded LDS row stride (shorts)

__global__ __launch_bounds__(256) void gemm_bias_kernel(
    const float* __restrict__ A, const float* __restrict__ W,
    const float* __restrict__ bias, float* __restrict__ C,
    int M, int N, int K)
{
  __shared__ unsigned short As[BM * LDT];   // [row][k]
  __shared__ unsigned short Bt[BN * LDT];   // [n][k]  (W transposed)

  const int tid  = threadIdx.x;
  const int lane = tid & 31;
  const int wave = tid >> 5;
  const int wm   = wave >> 2;               // 0..1
  const int wn   = wave & 3;                // 0..3
  const int row0 = blockIdx.y * BM;
  const int col0 = blockIdx.x * BN;

  const int Ml  = lane & 15;
  const int hhi = lane >> 4;
  const int kbA = hhi * 16;                 // A-frag K base (contiguous k = 16h + s)
  const int kbB = hhi * 16;                 // B-frag K base (contiguous k = 16h + s, same as A)

  CAcc acc[4][2];
  #pragma unroll
  for (int mt = 0; mt < 4; ++mt)
    #pragma unroll
    for (int nt = 0; nt < 2; ++nt)
      #pragma unroll
      for (int r = 0; r < 8; ++r) acc[mt][nt].f[r] = 0.0f;

  for (int k0 = 0; k0 < K; k0 += BK) {
    // stage A tile (128x32), fp32 -> bf16
    #pragma unroll
    for (int i = 0; i < 4; ++i) {
      const int idx = tid + i * 256;            // float4 slot 0..1023
      const int r   = idx >> 3;
      const int kc  = (idx & 7) << 2;
      const float4 a4 = *(const float4*)&A[(size_t)(row0 + r) * K + k0 + kc];
      unsigned short* p = &As[r * LDT + kc];
      p[0] = f2bf(a4.x); p[1] = f2bf(a4.y); p[2] = f2bf(a4.z); p[3] = f2bf(a4.w);
    }
    // stage W tile (32x128) transposed -> Bt[n][k]
    #pragma unroll
    for (int i = 0; i < 4; ++i) {
      const int idx = tid + i * 256;
      const int kr  = idx >> 5;
      const int nc  = (idx & 31) << 2;
      const float4 w4 = *(const float4*)&W[(size_t)(k0 + kr) * N + col0 + nc];
      Bt[(nc + 0) * LDT + kr] = f2bf(w4.x);
      Bt[(nc + 1) * LDT + kr] = f2bf(w4.y);
      Bt[(nc + 2) * LDT + kr] = f2bf(w4.z);
      Bt[(nc + 3) * LDT + kr] = f2bf(w4.w);
    }
    // prefetch next K-tile while this one is consumed (global_prefetch_b8)
    if (k0 + BK < K) {
      __builtin_prefetch(&A[(size_t)(row0 + (tid >> 3)) * K + k0 + BK + ((tid & 7) << 2)], 0, 1);
      __builtin_prefetch(&W[(size_t)(k0 + BK + (tid >> 5)) * N + col0 + ((tid & 31) << 2)], 0, 1);
    }
    __syncthreads();

    Frag a[4];
    #pragma unroll
    for (int mt = 0; mt < 4; ++mt) {
      const int r = wm * 64 + mt * 16 + Ml;
      #pragma unroll
      for (int v = 0; v < 8; ++v) {
        const int kk = kbA + 2 * v;
        a[mt].u[v] = *(const unsigned int*)&As[r * LDT + kk];
      }
    }
    Frag b[2];
    #pragma unroll
    for (int nt = 0; nt < 2; ++nt) {
      const int n = wn * 32 + nt * 16 + Ml;
      #pragma unroll
      for (int v = 0; v < 8; ++v)
        b[nt].u[v] = *(const unsigned int*)&Bt[n * LDT + kbB + 2 * v];
    }
    #pragma unroll
    for (int mt = 0; mt < 4; ++mt)
      #pragma unroll
      for (int nt = 0; nt < 2; ++nt)
        acc[mt][nt].v = __builtin_amdgcn_wmma_f32_16x16x32_bf16(
            false, a[mt].v, false, b[nt].v, (short)0, acc[mt][nt].v, false, false);
    __syncthreads();
  }

  #pragma unroll
  for (int mt = 0; mt < 4; ++mt)
    #pragma unroll
    for (int nt = 0; nt < 2; ++nt) {
      const int cc = col0 + wn * 32 + nt * 16 + Ml;
      const float bb = bias[cc];
      #pragma unroll
      for (int r = 0; r < 8; ++r) {
        const int rr = row0 + wm * 64 + mt * 16 + r + (hhi << 3);
        C[(size_t)rr * N + cc] = acc[mt][nt].f[r] + bb;
      }
    }
}

// ---------------------------------------------------------------------------
// RoPE + bf16 pack: Qf [S, H*D] -> Qh [H][S][D] bf16 (rotary applied);
// same for K; V just converted/transposed.
// ---------------------------------------------------------------------------
__global__ __launch_bounds__(256) void rope_pack_kernel(
    const float* __restrict__ Qf, const float* __restrict__ Kf,
    const float* __restrict__ Vf, const int* __restrict__ pos_ids,
    unsigned short* __restrict__ Qh, unsigned short* __restrict__ Kh,
    unsigned short* __restrict__ Vh)
{
  const int idx = blockIdx.x * 256 + threadIdx.x;
  if (idx >= S_LEN * N_HEADS * HEAD_DIM) return;
  const int d  = idx & (HEAD_DIM - 1);
  const int sh = idx >> 7;
  const int h  = sh & (N_HEADS - 1);
  const int s  = sh >> 5;

  const float pos = (float)pos_ids[s];
  const int dh = d & 63;
  // inv_freq = 10000^(-dh/64) = 2^(-dh * log2(10000)/64)
  const float ang = pos * exp2f(-(float)dh * (13.287712379549449f / 64.0f));
  float sn, cs;
  __sincosf(ang, &sn, &cs);

  const int qoff = s * (N_HEADS * HEAD_DIM) + h * HEAD_DIM;
  const float q  = Qf[qoff + d];
  const float qp = (d < 64) ? -Qf[qoff + d + 64] : Qf[qoff + d - 64];
  Qh[((size_t)h * S_LEN + s) * HEAD_DIM + d] = f2bf(q * cs + qp * sn);

  if (h < N_KV) {
    const int koff = s * (N_KV * HEAD_DIM) + h * HEAD_DIM;
    const float k  = Kf[koff + d];
    const float kp = (d < 64) ? -Kf[koff + d + 64] : Kf[koff + d - 64];
    Kh[((size_t)h * S_LEN + s) * HEAD_DIM + d] = f2bf(k * cs + kp * sn);
    Vh[((size_t)h * S_LEN + s) * HEAD_DIM + d] = f2bf(Vf[koff + d]);
  }
}

// ---------------------------------------------------------------------------
// Flash attention, causal, GQA (4 Q heads per KV head).
// Block = (128 q rows, 1 head); 8 waves, each owns a 16-row strip.
// K block staged via Tensor Data Mover (async DMA, TENSORcnt); V staged
// manually (needs transpose). Uniform trip count -> __syncthreads legal.
// ---------------------------------------------------------------------------
#define QT 128

__global__ __launch_bounds__(256) void attn_fwd_kernel(
    const unsigned short* __restrict__ Qh,
    const unsigned short* __restrict__ Kh,
    const unsigned short* __restrict__ Vh,
    float* __restrict__ O)
{
  __shared__ unsigned short Ks[32 * HEAD_DIM];   // [key][d]
  __shared__ unsigned short Vt[HEAD_DIM * 34];   // [d][key], padded
  __shared__ unsigned short Ps[8 * 16 * 34];     // per-wave P staging [16][34]

  const int h   = blockIdx.y;
  const int kvh = h >> 2;
  const int q0  = blockIdx.x * QT;
  const int tid  = threadIdx.x;
  const int lane = tid & 31;
  const int wave = tid >> 5;
  const int qr   = q0 + wave * 16;

  const int Ml  = lane & 15;
  const int hhi = lane >> 4;
  const int kbA = hhi * 16;
  const int kbB = hhi * 16;

  // Q fragments for this wave's 16 rows, K=128 -> 4 k-steps, kept in VGPRs
  Frag qf[4];
  {
    const size_t qbase = ((size_t)h * S_LEN + (qr + Ml)) * HEAD_DIM;
    #pragma unroll
    for (int kk = 0; kk < 4; ++kk)
      #pragma unroll
      for (int v = 0; v < 8; ++v) {
        const int dd = kk * 32 + kbA + 2 * v;
        qf[kk].u[v] = *(const unsigned int*)&Qh[qbase + dd];
      }
  }

  CAcc o[8];
  #pragma unroll
  for (int dt = 0; dt < 8; ++dt)
    #pragma unroll
    for (int r = 0; r < 8; ++r) o[dt].f[r] = 0.0f;
  float m_i[8], l_i[8];
  #pragma unroll
  for (int r = 0; r < 8; ++r) { m_i[r] = -1.0e30f; l_i[r] = 0.0f; }

  const float sc = 0.088388347648318447f * 1.4426950408889634f; // 1/sqrt(128) * log2(e)
  const unsigned ldsKs = (unsigned)(uintptr_t)(&Ks[0]);         // LDS byte offset

  const int kend = q0 + QT;                 // causal bound, uniform per block
  for (int kb0 = 0; kb0 < kend; kb0 += 32) {
    // ---- K block (32 keys x 128 dims, bf16) via TDM tensor_load_to_lds ----
    if (wave == 0) {
      const uint64_t ga =
          (uint64_t)(uintptr_t)(Kh + ((size_t)kvh * S_LEN + kb0) * HEAD_DIM);
      // D# group 0: count=1, lds_addr, global_addr[56:0], type=2
      v4u g0;
      g0[0] = 1u;
      g0[1] = ldsKs;
      g0[2] = (unsigned)ga;
      g0[3] = (unsigned)((ga >> 32) & 0x01FFFFFFu) | (2u << 30);
      // D# group 1: data_size=2B; tensor 128x32; tile 128x32; dim0 stride 128
      v8i g1;
      g1[0] = (int)(1u << 16);          // data_size=1 (2 bytes), wg_mask=0
      g1[1] = (int)(128u << 16);        // tensor_dim0[15:0]=128
      g1[2] = (int)(32u << 16);         // tensor_dim0 hi=0 | tensor_dim1[15:0]=32
      g1[3] = (int)(128u << 16);        // tensor_dim1 hi=0 | tile_dim0=128
      g1[4] = 32;                       // tile_dim1=32 | tile_dim2=0
      g1[5] = 128;                      // tensor_dim0_stride lo32 = 128
      g1[6] = 0;                        // stride0 hi | stride1 lo
      g1[7] = 0;                        // stride1 hi
      v4i gz; gz[0] = 0; gz[1] = 0; gz[2] = 0; gz[3] = 0;
#if defined(__clang_major__) && (__clang_major__ >= 23)
      v8i gz8;
      #pragma unroll
      for (int z = 0; z < 8; ++z) gz8[z] = 0;
      __builtin_amdgcn_tensor_load_to_lds(g0, g1, gz, gz, gz8, 0);
#else
      __builtin_amdgcn_tensor_load_to_lds(g0, g1, gz, gz, 0);
#endif
      __builtin_amdgcn_s_wait_tensorcnt(0);
    }
    // ---- V block transposed (manual; TDM cannot transpose): Vt[d][key] ----
    #pragma unroll
    for (int i = 0; i < 8; ++i) {
      const int idx = tid + i * 256;        // uint slot 0..2047
      const int key = idx >> 6;
      const int dd  = (idx & 63) << 1;
      const unsigned int vv =
          *(const unsigned int*)&Vh[((size_t)kvh * S_LEN + kb0 + key) * HEAD_DIM + dd];
      Vt[dd * 34 + key]       = (unsigned short)(vv & 0xFFFFu);
      Vt[(dd + 1) * 34 + key] = (unsigned short)(vv >> 16);
    }
    __syncthreads();

    // scores: two 16x16 tiles (keys kb0..+15, kb0+16..+31)
    CAcc s0, s1;
    #pragma unroll
    for (int r = 0; r < 8; ++r) { s0.f[r] = 0.0f; s1.f[r] = 0.0f; }
    #pragma unroll
    for (int kk = 0; kk < 4; ++kk) {
      Frag b0, b1;
      #pragma unroll
      for (int v = 0; v < 8; ++v) {
        const int dd = kk * 32 + kbB + 2 * v;
        b0.u[v] = *(const unsigned int*)&Ks[Ml * HEAD_DIM + dd];
        b1.u[v] = *(const unsigned int*)&Ks[(16 + Ml) * HEAD_DIM + dd];
      }
      s0.v = __builtin_amdgcn_wmma_f32_16x16x32_bf16(
          false, qf[kk].v, false, b0.v, (short)0, s0.v, false, false);
      s1.v = __builtin_amdgcn_wmma_f32_16x16x32_bf16(
          false, qf[kk].v, false, b1.v, (short)0, s1.v, false, false);
    }

    // online softmax (row reductions across each 16-lane half) + stage P
    unsigned short* pw = &Ps[wave * 16 * 34];
    #pragma unroll
    for (int r = 0; r < 8; ++r) {
      const int qrow = qr + r + (hhi << 3);
      const int key0 = kb0 + Ml;
      float x0 = (key0      <= qrow) ? s0.f[r] * sc : -1.0e30f;
      float x1 = (key0 + 16 <= qrow) ? s1.f[r] * sc : -1.0e30f;
      float mx = fmaxf(x0, x1);
      #pragma unroll
      for (int off = 8; off > 0; off >>= 1) mx = fmaxf(mx, __shfl_xor(mx, off, 32));
      const float mnew = fmaxf(m_i[r], mx);
      const float corr = exp2f(m_i[r] - mnew);
      const float p0 = exp2f(x0 - mnew);
      const float p1 = exp2f(x1 - mnew);
      float rs = p0 + p1;
      #pragma unroll
      for (int off = 8; off > 0; off >>= 1) rs += __shfl_xor(rs, off, 32);
      l_i[r] = l_i[r] * corr + rs;
      m_i[r] = mnew;
      const int lrow = r + (hhi << 3);
      pw[lrow * 34 + Ml]      = f2bf(p0);
      pw[lrow * 34 + 16 + Ml] = f2bf(p1);
      #pragma unroll
      for (int dt = 0; dt < 8; ++dt) o[dt].f[r] *= corr;
    }
    // wave-private LDS RAW: ensure P stores landed before fragment reads
    asm volatile("s_wait_dscnt 0x0" ::: "memory");

    // P fragment (A-matrix layout, 16x32)
    Frag pa;
    #pragma unroll
    for (int v = 0; v < 8; ++v) {
      const int kk = kbA + 2 * v;
      pa.u[v] = *(const unsigned int*)&pw[Ml * 34 + kk];
    }
    // P @ V : 8 output d-tiles
    #pragma unroll
    for (int dt = 0; dt < 8; ++dt) {
      Frag vb;
      #pragma unroll
      for (int v = 0; v < 8; ++v)
        vb.u[v] = *(const unsigned int*)&Vt[(dt * 16 + Ml) * 34 + kbB + 2 * v];
      o[dt].v = __builtin_amdgcn_wmma_f32_16x16x32_bf16(
          false, pa.v, false, vb.v, (short)0, o[dt].v, false, false);
    }
    __syncthreads();
  }

  // normalize and write attn output [S, H*D] fp32
  #pragma unroll
  for (int dt = 0; dt < 8; ++dt)
    #pragma unroll
    for (int r = 0; r < 8; ++r) {
      const int qrow = qr + r + (hhi << 3);
      const int dd = dt * 16 + Ml;
      O[(size_t)qrow * (N_HEADS * HEAD_DIM) + h * HEAD_DIM + dd] = o[dt].f[r] / l_i[r];
    }
}

// ---------------------------------------------------------------------------
// Host side
// ---------------------------------------------------------------------------
extern "C" void kernel_launch(void* const* d_in, const int* in_sizes, int n_in,
                              void* d_out, int out_size, void* d_ws, size_t ws_size,
                              hipStream_t stream)
{
  (void)in_sizes; (void)n_in; (void)out_size; (void)ws_size;
  const float* X  = (const float*)d_in[0];
  // d_in[1]: attention_mask (all-true padding mask); causal handled in-kernel
  const int*   pos = (const int*)d_in[2];
  const float* Wq = (const float*)d_in[3];
  const float* bq = (const float*)d_in[4];
  const float* Wk = (const float*)d_in[5];
  const float* bk = (const float*)d_in[6];
  const float* Wv = (const float*)d_in[7];
  const float* bv = (const float*)d_in[8];
  const float* Wo = (const float*)d_in[9];
  const float* bo = (const float*)d_in[10];
  float* out = (float*)d_out;

  char* ws = (char*)d_ws;
  float* Qf = (float*)ws;                                               // [S, 4096] fp32
  float* Kf = Qf + (size_t)S_LEN * (N_HEADS * HEAD_DIM);                // [S, 1024]
  float* Vf = Kf + (size_t)S_LEN * (N_KV * HEAD_DIM);                   // [S, 1024]
  unsigned short* Qh = (unsigned short*)(Vf + (size_t)S_LEN * (N_KV * HEAD_DIM)); // [H][S][D] bf16
  unsigned short* Kh = Qh + (size_t)S_LEN * N_HEADS * HEAD_DIM;         // [KV][S][D]
  unsigned short* Vh = Kh + (size_t)S_LEN * N_KV * HEAD_DIM;            // [KV][S][D]
  float* AO = Qf;   // Qf dead after rope_pack -> reuse for attn output [S, 4096]

  dim3 blk(256);
  gemm_bias_kernel<<<dim3(4096 / BN, S_LEN / BM), blk, 0, stream>>>(X, Wq, bq, Qf, S_LEN, 4096, HIDDEN);
  gemm_bias_kernel<<<dim3(1024 / BN, S_LEN / BM), blk, 0, stream>>>(X, Wk, bk, Kf, S_LEN, 1024, HIDDEN);
  gemm_bias_kernel<<<dim3(1024 / BN, S_LEN / BM), blk, 0, stream>>>(X, Wv, bv, Vf, S_LEN, 1024, HIDDEN);
  rope_pack_kernel<<<dim3((S_LEN * N_HEADS * HEAD_DIM) / 256), blk, 0, stream>>>(
      Qf, Kf, Vf, pos, Qh, Kh, Vh);
  attn_fwd_kernel<<<dim3(S_LEN / QT, N_HEADS), blk, 0, stream>>>(Qh, Kh, Vh, AO);
  gemm_bias_kernel<<<dim3(4096 / BN, S_LEN / BM), blk, 0, stream>>>(AO, Wo, bo, out, S_LEN, 4096, HIDDEN);
}
